// Neuralsymbol_10368051052621
// MI455X (gfx1250) — compile-verified, matrix operand order checked by reading, not run on hardware
//
#include <hip/hip_runtime.h>
#include <hip/hip_bf16.h>

// ---------------- constants from the reference ----------------
#define B_   256
#define T_   128
#define ID_  256
#define HD_  256
#define OD_  256
#define NQ_  4
#define NS_  2
#define HH_  64
#define K3_  768          // 3*ID
#define VS_  772          // padded f32 row stride for concat vector (768 + 4)
#define HS_  68           // padded f32 row stride for hh (64 + 4)
#define ZO_  1e-3f

typedef __attribute__((ext_vector_type(16))) __bf16 v16bf;
typedef __attribute__((ext_vector_type(8)))  float  v8f;
typedef __attribute__((ext_vector_type(4)))  unsigned int u32x4;   // plain vector (not HIP class)

// Global (addrspace(1)) pointer views to force global_load/store instead of flat.
typedef __attribute__((address_space(1))) const u32x4* gcu4;
typedef __attribute__((address_space(1))) const float* gcf;
typedef __attribute__((address_space(1))) float*       gf;

union ABf { unsigned int u[8]; v16bf v; };

__device__ __forceinline__ unsigned short f2bf(float f) {
  unsigned int u = __float_as_uint(f);
  u += 0x7FFFu + ((u >> 16) & 1u);      // round-to-nearest-even
  return (unsigned short)(u >> 16);
}

__device__ __forceinline__ float gelu_exact(float x) {
  return 0.5f * x * (1.0f + erff(x * 0.70710678118654752f));
}
__device__ __forceinline__ float sigmoidf_(float x) {
  return 1.0f / (1.0f + expf(-x));
}

// A fragment (16x32 bf16, M x K): lanes 0-15 row M=lane, K {0-7,16-23};
// lanes 16-31 row M=lane-16, K {8-15,24-31}. Built from fp32 LDS row.
__device__ __forceinline__ v16bf load_a_frag(const float* row, int kbase) {
  ABf a;
#pragma unroll
  for (int j = 0; j < 4; ++j) {
    unsigned int lo = f2bf(row[kbase + 2 * j]);
    unsigned int hi = f2bf(row[kbase + 2 * j + 1]);
    a.u[j] = lo | (hi << 16);
  }
#pragma unroll
  for (int j = 0; j < 4; ++j) {
    unsigned int lo = f2bf(row[kbase + 16 + 2 * j]);
    unsigned int hi = f2bf(row[kbase + 16 + 2 * j + 1]);
    a.u[4 + j] = lo | (hi << 16);
  }
  return a.v;
}

// B fragment: pre-packed tile of 32 lanes x 16 bf16 (32 B/lane, contiguous).
// Forced to addrspace(1) => global_load_b128 (LOADcnt only, no DScnt).
__device__ __forceinline__ v16bf load_b_frag(const unsigned short* tile, int lane) {
  gcu4 p = (gcu4)(unsigned long long)(tile + lane * 16);
  u32x4 x0 = p[0];
  u32x4 x1 = p[1];
  ABf b;
  b.u[0] = x0[0]; b.u[1] = x0[1]; b.u[2] = x0[2]; b.u[3] = x0[3];
  b.u[4] = x1[0]; b.u[5] = x1[1]; b.u[6] = x1[2]; b.u[7] = x1[3];
  return b.v;
}

__device__ __forceinline__ float gload(const float* p) {
  return *(gcf)(unsigned long long)p;
}

// ---------------- weight pre-pack: [g][K][Npg] f32 -> WMMA B tiles bf16 -----
__global__ void pack_wmma_b(const float* __restrict__ src, unsigned short* __restrict__ dst,
                            int Ntiles, int Ktiles, int Npg, int K) {
  long e = (long)blockIdx.x * blockDim.x + threadIdx.x;
  long total = (long)Ntiles * Ktiles * 512;
  if (e >= total) return;
  int within = (int)(e & 511);
  long tile  = e >> 9;
  int kt = (int)(tile % Ktiles);
  int nt = (int)(tile / Ktiles);
  int lane = within >> 4, j = within & 15;
  int nl = lane & 15;
  int kb = (lane >> 4) * 8;
  int k  = kt * 32 + kb + j + ((j >= 8) ? 8 : 0);   // mirror of A-fragment K map
  int cg = nt * 16 + nl;
  int g  = cg / Npg, h = cg % Npg;
  dst[e] = f2bf(src[((long)g * K + k) * Npg + h]);
}

// ---------------- fused gate MLP pass (6 gates, K=768, N=1536) --------------
__device__ void gate_pass(const float* sv, const unsigned short* pw,
                          const float* __restrict__ b1, const float* __restrict__ w2,
                          float* s_logits, int wave, int lane) {
  const int m  = lane & 15;
  const int kb = (lane >> 4) * 8;
  const int mbase = (lane >> 4) * 8;
  const float* row = sv + m * VS_;
  for (int nt = wave; nt < 96; nt += 8) {
    int g = nt >> 4;
    const unsigned short* tb = pw + (size_t)nt * 24 * 512;
    v8f acc = {0.f, 0.f, 0.f, 0.f, 0.f, 0.f, 0.f, 0.f};
    // software pipeline: B(kt+1) in flight + prefetch B(kt+2) under WMMA(kt)
    v16bf bcur = load_b_frag(tb, lane);
    for (int kt = 0; kt < 24; ++kt) {
      v16bf bnext = bcur;
      if (kt + 1 < 24) bnext = load_b_frag(tb + (kt + 1) * 512, lane);
      if (kt + 2 < 24) __builtin_prefetch(tb + (kt + 2) * 512 + lane * 16, 0, 0);
      v16bf a = load_a_frag(row, kt * 32 + kb);
      acc = __builtin_amdgcn_wmma_f32_16x16x32_bf16(false, a, false, bcur,
                                                    (short)0, acc, false, false);
      bcur = bnext;
    }
    int h = ((nt * 16) & 255) + m;            // column within this gate's HD
    float bias = gload(&b1[g * HD_ + h]);
    float wv   = gload(&w2[g * HD_ + h]);
#pragma unroll
    for (int r = 0; r < 8; ++r) {
      float v = gelu_exact(acc[r] + bias) * wv;
      v += __shfl_xor(v, 1);
      v += __shfl_xor(v, 2);
      v += __shfl_xor(v, 4);
      v += __shfl_xor(v, 8);
      if (m == 0) atomicAdd(&s_logits[g * 16 + mbase + r], v);
    }
  }
}

// ---------------- ff stage 1: xx[16,768] @ ffW1 -> gelu -> hh[16,64] --------
__device__ void ff1_pass(const float* sv, const unsigned short* pw, float* s_hh,
                         int wave, int lane) {
  if (wave >= 4) return;
  const int nt = wave;
  const int m  = lane & 15;
  const int kb = (lane >> 4) * 8;
  const int mbase = (lane >> 4) * 8;
  const float* row = sv + m * VS_;
  const unsigned short* tb = pw + (size_t)nt * 24 * 512;
  v8f acc = {0.f, 0.f, 0.f, 0.f, 0.f, 0.f, 0.f, 0.f};
  v16bf bcur = load_b_frag(tb, lane);
  for (int kt = 0; kt < 24; ++kt) {
    v16bf bnext = bcur;
    if (kt + 1 < 24) bnext = load_b_frag(tb + (kt + 1) * 512, lane);
    v16bf a = load_a_frag(row, kt * 32 + kb);
    acc = __builtin_amdgcn_wmma_f32_16x16x32_bf16(false, a, false, bcur,
                                                  (short)0, acc, false, false);
    bcur = bnext;
  }
  int col = nt * 16 + m;
#pragma unroll
  for (int r = 0; r < 8; ++r)
    s_hh[(mbase + r) * HS_ + col] = gelu_exact(acc[r]);
}

// ---------------- ff stage 2: hh @ ffW2 -> tanh -> out ----------------------
__device__ void ff2_pass(const float* s_hh, const unsigned short* pw, float* sv,
                         float* __restrict__ outp, int t, int row0, int wave, int lane) {
  const int m  = lane & 15;
  const int kb = (lane >> 4) * 8;
  const int mbase = (lane >> 4) * 8;
  const float* row = s_hh + m * HS_;
  gf outg = (gf)(unsigned long long)outp;
#pragma unroll
  for (int i = 0; i < 2; ++i) {
    int nt = wave + i * 8;
    const unsigned short* tb = pw + (size_t)nt * 2 * 512;
    v8f acc = {0.f, 0.f, 0.f, 0.f, 0.f, 0.f, 0.f, 0.f};
#pragma unroll
    for (int kt = 0; kt < 2; ++kt) {
      v16bf a = load_a_frag(row, kt * 32 + kb);
      v16bf b = load_b_frag(tb + kt * 512, lane);
      acc = __builtin_amdgcn_wmma_f32_16x16x32_bf16(false, a, false, b,
                                                    (short)0, acc, false, false);
    }
    int col = nt * 16 + m;
#pragma unroll
    for (int r = 0; r < 8; ++r) {
      int M = mbase + r;
      float o = tanhf(acc[r]);
      sv[M * VS_ + col] = o;  // out becomes segment 0 of the next concat
      outg[((size_t)(row0 + M) * T_ + t) * OD_ + col] = o;
    }
  }
}

// ---------------- soft queue / stack state updates --------------------------
__device__ void queue_cells(float* q, const float* head, const float* tail,
                            const float* gates, const float* value, int tid) {
  for (int idx = tid; idx < 16 * 256; idx += 256) {
    int r = idx >> 8, c = idx & 255;
    float put = gates[0 * 16 + r], get = gates[1 * 16 + r], nop = gates[2 * 16 + r];
    float inv = 1.0f / (put + get + nop + 1e-6f);
    float v = value[r * VS_ + c];
#pragma unroll
    for (int n = 0; n < NQ_; ++n) {
      float qq = q[(r * NQ_ + n) * 256 + c];
      float h  = head[r * NQ_ + n];
      float tl = tail[r * NQ_ + n];
      float qp = qq * (1.f - h)  + h  * v;
      float qg = qq * (1.f - tl) + tl * ZO_;
      q[(r * NQ_ + n) * 256 + c] = (put * qp + get * qg + nop * qq) * inv;
    }
  }
}

__device__ void queue_ptrs(const float* head, const float* tail,
                           float* head2, float* tail2,
                           const float* gates, float shH, float shT, int r) {
  float put = gates[r], get = gates[16 + r], nop = gates[32 + r];
  float inv = 1.f / (put + get + nop + 1e-6f);
  float h[NQ_], tl[NQ_], nh[NQ_], ntl[NQ_];
#pragma unroll
  for (int n = 0; n < NQ_; ++n) { h[n] = head[r * NQ_ + n]; tl[n] = tail[r * NQ_ + n]; }
#pragma unroll
  for (int n = 0; n < NQ_; ++n)
    nh[n] = (put * h[(n + 3) & 3] + (get + nop) * h[n]) * inv;
  float s = 0.f;
#pragma unroll
  for (int n = 0; n < NQ_; ++n) { nh[n] = powf(fmaxf(nh[n], 1e-10f), shH); s += nh[n]; }
  s = 1.f / s;
#pragma unroll
  for (int n = 0; n < NQ_; ++n) head2[r * NQ_ + n] = nh[n] * s;
#pragma unroll
  for (int n = 0; n < NQ_; ++n)
    ntl[n] = (get * tl[(n + 3) & 3] + (put + nop) * tl[n]) * inv;
  s = 0.f;
#pragma unroll
  for (int n = 0; n < NQ_; ++n) { ntl[n] = powf(fmaxf(ntl[n], 1e-10f), shT); s += ntl[n]; }
  s = 1.f / s;
#pragma unroll
  for (int n = 0; n < NQ_; ++n) tail2[r * NQ_ + n] = ntl[n] * s;
}

__device__ void stack_cells(float* st, const float* ptr,
                            const float* gates, const float* value, int tid) {
  for (int idx = tid; idx < 16 * 256; idx += 256) {
    int r = idx >> 8, c = idx & 255;
    float push = gates[3 * 16 + r], pop = gates[4 * 16 + r], nop = gates[5 * 16 + r];
    float inv = 1.0f / (push + pop + nop + 1e-6f);
    float v = value[r * VS_ + c];
#pragma unroll
    for (int n = 0; n < NS_; ++n) {
      float s  = st[(r * NS_ + n) * 256 + c];
      float pp = ptr[r * NS_ + ((n + NS_ - 1) % NS_)];   // roll(ptr, 1)
      float po = ptr[r * NS_ + n];
      float sp = s * (1.f - pp) + pp * v;
      float sg = s * (1.f - po) + po * ZO_;
      st[(r * NS_ + n) * 256 + c] = (push * sp + pop * sg + nop * s) * inv;
    }
  }
}

__device__ void stack_ptrs(const float* ptr, float* ptr2,
                           const float* gates, float shS, int r) {
  float push = gates[3 * 16 + r], pop = gates[4 * 16 + r], nop = gates[5 * 16 + r];
  float inv = 1.f / (push + pop + nop + 1e-6f);
  float p[NS_], np[NS_];
#pragma unroll
  for (int n = 0; n < NS_; ++n) p[n] = ptr[r * NS_ + n];
#pragma unroll
  for (int n = 0; n < NS_; ++n)
    np[n] = (push * p[(n + NS_ - 1) % NS_] + pop * p[(n + 1) % NS_] + nop * p[n]) * inv;
  float s = 0.f;
#pragma unroll
  for (int n = 0; n < NS_; ++n) { np[n] = powf(fmaxf(np[n], 1e-10f), shS); s += np[n]; }
  s = 1.f / s;
#pragma unroll
  for (int n = 0; n < NS_; ++n) ptr2[r * NS_ + n] = np[n] * s;
}

__device__ void readouts(const float* q, const float* tail,
                         const float* st, const float* ptr, float* sv, int tid) {
  for (int idx = tid; idx < 16 * 256; idx += 256) {
    int r = idx >> 8, c = idx & 255;
    float a = 0.f;
#pragma unroll
    for (int n = 0; n < NQ_; ++n) a += tail[r * NQ_ + n] * q[(r * NQ_ + n) * 256 + c];
    sv[r * VS_ + 256 + c] = a;
    float b = 0.f;
#pragma unroll
    for (int n = 0; n < NS_; ++n) b += ptr[r * NS_ + n] * st[(r * NS_ + n) * 256 + c];
    sv[r * VS_ + 512 + c] = b;
  }
}

// ---------------- persistent recurrent kernel: 1 block = 16 batch rows ------
__global__ __launch_bounds__(256) void ns_persistent(
    const float* __restrict__ x,
    const unsigned short* __restrict__ pwA,
    const unsigned short* __restrict__ pwB,
    const unsigned short* __restrict__ pwF1,
    const unsigned short* __restrict__ pwF2,
    const float* __restrict__ ba1, const float* __restrict__ Wa2, const float* __restrict__ ba2,
    const float* __restrict__ bb1, const float* __restrict__ Wb2, const float* __restrict__ bb2,
    const float* __restrict__ shHp, const float* __restrict__ shTp, const float* __restrict__ shSp,
    float* __restrict__ outp) {
  __shared__ float s_vec[16 * VS_];            // concat [inp|q read|stack read]
  __shared__ float s_queue[16 * NQ_ * 256];
  __shared__ float s_stack[16 * NS_ * 256];
  __shared__ float s_head[64], s_tail[64], s_head2[64], s_tail2[64];
  __shared__ float s_ptr[32], s_ptr2[32];
  __shared__ float s_hh[16 * HS_];
  __shared__ float s_logits[96];
  __shared__ float s_gates[96];

  const int tid  = threadIdx.x;
  const int wave = tid >> 5;
  const int lane = tid & 31;
  const int row0 = blockIdx.x * 16;
  const float shH = gload(&shHp[0]), shT = gload(&shTp[0]), shS = gload(&shSp[0]);

  // init recurrent state
  for (int i = tid; i < 16 * NQ_ * 256; i += 256) s_queue[i] = ZO_;
  for (int i = tid; i < 16 * NS_ * 256; i += 256) s_stack[i] = ZO_;
  if (tid < 64) { int n = tid & 3; s_head[tid] = (n == 1) ? 1.f : 0.f;
                  s_tail[tid] = (n == 0) ? 1.f : 0.f; }
  if (tid < 32) s_ptr[tid] = ((tid & 1) == 0) ? 1.f : 0.f;
  __syncthreads();

  for (int t = 0; t < T_; ++t) {
    // ---- async DMA the x tile for this step straight into LDS (ASYNCcnt) ----
    // 16 rows x 256 f32 = 1024 chunks of 16B; 4 per thread.
    for (int idx = tid; idx < 16 * 64; idx += 256) {
      int r = idx >> 6, c4 = (idx & 63) * 4;
      unsigned ldsaddr = (unsigned)(unsigned long long)&s_vec[r * VS_ + c4];
      const float* ga = x + ((size_t)(row0 + r) * T_ + t) * ID_ + c4;
      asm volatile("global_load_async_to_lds_b128 %0, %1, off"
                   :: "v"(ldsaddr), "v"(ga) : "memory");
    }
    // ---- meanwhile compute q1h / srd readouts on the VALU ----
    for (int idx = tid; idx < 16 * 256; idx += 256) {
      int r = idx >> 8, c = idx & 255;
      float a = 0.f;
#pragma unroll
      for (int n = 0; n < NQ_; ++n) a += s_tail[r * NQ_ + n] * s_queue[(r * NQ_ + n) * 256 + c];
      s_vec[r * VS_ + 256 + c] = a;
      float b = 0.f;
#pragma unroll
      for (int n = 0; n < NS_; ++n) b += s_ptr[r * NS_ + n] * s_stack[(r * NS_ + n) * 256 + c];
      s_vec[r * VS_ + 512 + c] = b;
    }
    if (tid < 96) s_logits[tid] = 0.f;
    asm volatile("s_wait_asynccnt 0x0" ::: "memory");
    __syncthreads();

    // ---- group-a gates ----
    gate_pass(s_vec, pwA, ba1, Wa2, s_logits, wave, lane);
    __syncthreads();
    if (tid < 96) s_gates[tid] = sigmoidf_(s_logits[tid] + gload(&ba2[tid >> 4]));
    __syncthreads();

    // ---- queue/stack step a (value = inp) ----
    queue_cells(s_queue, s_head, s_tail, s_gates, s_vec, tid);
    stack_cells(s_stack, s_ptr, s_gates, s_vec, tid);
    if (tid < 16) {
      queue_ptrs(s_head, s_tail, s_head2, s_tail2, s_gates, shH, shT, tid);
      stack_ptrs(s_ptr, s_ptr2, s_gates, shS, tid);
    }
    __syncthreads();
    if (tid < 64) { s_head[tid] = s_head2[tid]; s_tail[tid] = s_tail2[tid]; }
    if (tid < 32) s_ptr[tid] = s_ptr2[tid];
    __syncthreads();
    readouts(s_queue, s_tail, s_stack, s_ptr, s_vec, tid);   // q1x, sx
    __syncthreads();

    // ---- output FF: xx -> gelu -> tanh ----
    ff1_pass(s_vec, pwF1, s_hh, wave, lane);
    __syncthreads();
    ff2_pass(s_hh, pwF2, s_vec, outp, t, row0, wave, lane);  // writes out segment
    if (tid < 96) s_logits[tid] = 0.f;
    __syncthreads();

    // ---- group-b gates on b = [out, q1x, sx] ----
    gate_pass(s_vec, pwB, bb1, Wb2, s_logits, wave, lane);
    __syncthreads();
    if (tid < 96) s_gates[tid] = sigmoidf_(s_logits[tid] + gload(&bb2[tid >> 4]));
    __syncthreads();

    // ---- queue/stack step b (value = out) ----
    queue_cells(s_queue, s_head, s_tail, s_gates, s_vec, tid);
    stack_cells(s_stack, s_ptr, s_gates, s_vec, tid);
    if (tid < 16) {
      queue_ptrs(s_head, s_tail, s_head2, s_tail2, s_gates, shH, shT, tid);
      stack_ptrs(s_ptr, s_ptr2, s_gates, shS, tid);
    }
    __syncthreads();
    if (tid < 64) { s_head[tid] = s_head2[tid]; s_tail[tid] = s_tail2[tid]; }
    if (tid < 32) s_ptr[tid] = s_ptr2[tid];
    __syncthreads();
  }
}

// ---------------- host-side launch ------------------------------------------
extern "C" void kernel_launch(void* const* d_in, const int* in_sizes, int n_in,
                              void* d_out, int out_size, void* d_ws, size_t ws_size,
                              hipStream_t stream) {
  const float* x    = (const float*)d_in[0];
  const float* Wa1  = (const float*)d_in[1];
  const float* ba1  = (const float*)d_in[2];
  const float* Wa2  = (const float*)d_in[3];
  const float* ba2  = (const float*)d_in[4];
  const float* Wb1  = (const float*)d_in[5];
  const float* bb1  = (const float*)d_in[6];
  const float* Wb2  = (const float*)d_in[7];
  const float* bb2  = (const float*)d_in[8];
  const float* ffW1 = (const float*)d_in[9];
  const float* ffW2 = (const float*)d_in[10];
  const float* shH  = (const float*)d_in[11];
  const float* shT  = (const float*)d_in[12];
  const float* shS  = (const float*)d_in[13];
  float* out = (float*)d_out;

  unsigned short* ws = (unsigned short*)d_ws;
  const long nA  = 96L * 24 * 512;   // gate weight tiles (per group)
  const long nF1 = 4L  * 24 * 512;
  const long nF2 = 16L * 2  * 512;
  unsigned short* pA  = ws;
  unsigned short* pB  = pA  + nA;
  unsigned short* pF1 = pB  + nA;
  unsigned short* pF2 = pF1 + nF1;

  const int thr = 256;
  pack_wmma_b<<<(int)((nA  + thr - 1) / thr), thr, 0, stream>>>(Wa1,  pA,  96, 24, 256, 768);
  pack_wmma_b<<<(int)((nA  + thr - 1) / thr), thr, 0, stream>>>(Wb1,  pB,  96, 24, 256, 768);
  pack_wmma_b<<<(int)((nF1 + thr - 1) / thr), thr, 0, stream>>>(ffW1, pF1,  4, 24,  64, 768);
  pack_wmma_b<<<(int)((nF2 + thr - 1) / thr), thr, 0, stream>>>(ffW2, pF2, 16,  2, 256,  64);

  ns_persistent<<<16, 256, 0, stream>>>(x, pA, pB, pF1, pF2,
                                        ba1, Wa2, ba2, bb1, Wb2, bb2,
                                        shH, shT, shS, out);
}
